// MultiHeadAttention_44521630990959
// MI455X (gfx1250) — compile-verified, matrix operand order checked by reading, not run on hardware
//
#include <hip/hip_runtime.h>
#include <hip/hip_bf16.h>

// ---- problem constants (match reference) ----
#define HH   16
#define DM   1024
#define DKV  64
#define NBAT 4
#define LSEQ 2048
#define NL   (NBAT * LSEQ)

typedef __attribute__((ext_vector_type(16))) __bf16 v16bf;
typedef __attribute__((ext_vector_type(8)))  float  v8f;
typedef __attribute__((ext_vector_type(4)))  unsigned int uint4v;
typedef __attribute__((ext_vector_type(4)))  int int4v;

union Frag { v16bf bf; uint4v u[2]; };

#define WMMA_BF16(a, b, c) \
  __builtin_amdgcn_wmma_f32_16x16x32_bf16(false, (a), false, (b), (short)0, (c), false, false)

// CDNA5 async global->LDS path (ASYNCcnt-tracked), guarded so either toolchain compiles.
#if defined(__gfx1250__) && __has_builtin(__builtin_amdgcn_global_load_async_to_lds_b128) && \
    __has_builtin(__builtin_amdgcn_s_wait_asynccnt)
#define HAS_ASYNC_LDS 1
#endif

__device__ __forceinline__ void cp16_g2lds(unsigned short* ldst, const unsigned short* gsrc) {
#ifdef HAS_ASYNC_LDS
  __builtin_amdgcn_global_load_async_to_lds_b128(
      (__attribute__((address_space(1))) int4v*)(gsrc),
      (__attribute__((address_space(3))) int4v*)(ldst), 0, 0);
#else
  *(uint4v*)ldst = *(const uint4v*)gsrc;
#endif
}
__device__ __forceinline__ void cp_g2lds_wait() {
#ifdef HAS_ASYNC_LDS
  __builtin_amdgcn_s_wait_asynccnt(0);
#endif
}

__device__ __forceinline__ unsigned short f2bf(float f) {
  unsigned int u = __float_as_uint(f);
  u += 0x7FFFu + ((u >> 16) & 1u);   // round-to-nearest-even
  return (unsigned short)(u >> 16);
}

// Load one 16x32 (A) or 32x16 (B) bf16 fragment.
// ISA 7.12.2 (16-bit operands, wave32): lane = outer(0..15) + 16*half;
// regs hold K = {8h..8h+7} and {16+8h..16+8h+7} -> two contiguous 16B runs.
__device__ __forceinline__ v16bf load_frag(const unsigned short* base, int stride, int lane) {
  int r = lane & 15, h = lane >> 4;
  Frag f;
  f.u[0] = *(const uint4v*)(base + r * stride + 8 * h);
  f.u[1] = *(const uint4v*)(base + r * stride + 16 + 8 * h);
  return f.bf;
}

// ------------------- conversion kernels -------------------
__global__ void k_conv_x(const float* __restrict__ x, unsigned short* __restrict__ xb, int n) {
  int i = blockIdx.x * blockDim.x + threadIdx.x;
  if (i < n) xb[i] = f2bf(x[i]);
}

// W: [H][DM][DKV] -> Wt: [H][DKV][DM]  (columns contiguous for WMMA B)
__global__ void k_conv_w(const float* __restrict__ w, unsigned short* __restrict__ wt) {
  int i = blockIdx.x * blockDim.x + threadIdx.x;    // < HH*DM*DKV
  int h = i / (DM * DKV);
  int rem = i % (DM * DKV);
  int d = rem / DKV;
  int c = rem % DKV;
  wt[(h * DKV + c) * DM + d] = f2bf(w[i]);
}

// WO: [DM(in)][DM(out)] -> WOt: [out][in]
__global__ void k_conv_wo(const float* __restrict__ wo, unsigned short* __restrict__ wot) {
  int i = blockIdx.x * blockDim.x + threadIdx.x;    // < DM*DM
  int in = i / DM, out = i % DM;
  wot[out * DM + in] = f2bf(wo[i]);
}

// ------------------- QKV projection -------------------
// grid.x = NL/128, grid.y = 3*HH; block = 128 (4 waves).
// Each wave: 32 rows x 64 cols (whole head width) => 8 WMMAs per 6 frag loads.
__global__ void __launch_bounds__(128) k_qkv(
    const unsigned short* __restrict__ xb,
    const unsigned short* __restrict__ wqt,
    const unsigned short* __restrict__ wkt,
    const unsigned short* __restrict__ wvt,
    unsigned short* __restrict__ q,
    unsigned short* __restrict__ k,
    unsigned short* __restrict__ vt) {
  int mt    = blockIdx.x;
  int which = blockIdx.y / HH;
  int h     = blockIdx.y % HH;
  int wave  = threadIdx.x >> 5;
  int lane  = threadIdx.x & 31;

  const unsigned short* wt = (which == 0) ? wqt : ((which == 1) ? wkt : wvt);
  const unsigned short* Bbase = wt + (size_t)h * DKV * DM;               // 64 cols, stride DM
  const unsigned short* Abase = xb + (size_t)(mt * 128 + wave * 32) * DM;

  v8f acc[2][4] = {};
  for (int kk = 0; kk < DM; kk += 32) {
    __builtin_prefetch(Abase + kk + 256, 0, 1);
    __builtin_prefetch(Bbase + kk + 256, 0, 1);
    v16bf a0 = load_frag(Abase + kk, DM, lane);
    v16bf a1 = load_frag(Abase + (size_t)16 * DM + kk, DM, lane);
#pragma unroll
    for (int t = 0; t < 4; ++t) {
      v16bf b = load_frag(Bbase + (size_t)(t * 16) * DM + kk, DM, lane);
      acc[0][t] = WMMA_BF16(a0, b, acc[0][t]);
      acc[1][t] = WMMA_BF16(a1, b, acc[1][t]);
    }
  }

  int c  = lane & 15, hf = lane >> 4;
  int n  = (mt * 128) / LSEQ;          // 128-row tiles never straddle batches
#pragma unroll
  for (int rt = 0; rt < 2; ++rt) {
    int l0 = (mt * 128 + wave * 32 + rt * 16 + 8 * hf) % LSEQ;
    if (which < 2) {
      unsigned short* dst = (which == 0) ? q : k;   // [h][n][l][DKV]
      size_t base = (size_t)(h * NBAT + n) * LSEQ * DKV;
#pragma unroll
      for (int t = 0; t < 4; ++t)
#pragma unroll
        for (int j = 0; j < 8; ++j)
          dst[base + (size_t)(l0 + j) * DKV + t * 16 + c] = f2bf(acc[rt][t][j]);
    } else {
      // vt: [h][n][dv][L] (rows contiguous -> packed b128 store)
#pragma unroll
      for (int t = 0; t < 4; ++t) {
        unsigned short tmp[8];
#pragma unroll
        for (int j = 0; j < 8; ++j) tmp[j] = f2bf(acc[rt][t][j]);
        uint4v pack;
        pack.x = (unsigned)tmp[0] | ((unsigned)tmp[1] << 16);
        pack.y = (unsigned)tmp[2] | ((unsigned)tmp[3] << 16);
        pack.z = (unsigned)tmp[4] | ((unsigned)tmp[5] << 16);
        pack.w = (unsigned)tmp[6] | ((unsigned)tmp[7] << 16);
        *(uint4v*)(vt + ((size_t)(h * NBAT + n) * DKV + t * 16 + c) * LSEQ + l0) = pack;
      }
    }
  }
}

// ------------------- flash attention -------------------
// grid.x = NL/128, grid.y = HH; block = 256 (8 waves, each owns a 16-query tile)
__global__ void __launch_bounds__(256) k_attn(
    const unsigned short* __restrict__ q,
    const unsigned short* __restrict__ k,
    const unsigned short* __restrict__ vt,
    const float* __restrict__ mi,
    unsigned short* __restrict__ ctx) {
  __shared__ unsigned short Kc[32 * 64];        // [key][dk]
  __shared__ unsigned short Vc[64 * 32];        // [dv][key]
  __shared__ unsigned short Pst[8][16 * 32];    // per-wave P transpose buffer

  int h    = blockIdx.y;
  int tile = blockIdx.x;
  int n    = tile / (LSEQ / 128);
  int qb   = (tile % (LSEQ / 128)) * 128;
  int tid  = threadIdx.x;
  int wave = tid >> 5;
  int lane = tid & 31;
  int c    = lane & 15, hf = lane >> 4;

  size_t hn = (size_t)(h * NBAT + n);
  const unsigned short* qbase = q  + (hn * LSEQ + qb + wave * 16) * DKV;
  const unsigned short* kbase = k  + hn * LSEQ * DKV;
  const unsigned short* vbase = vt + hn * DKV * LSEQ;
  const float* mrow = mi + (size_t)n * LSEQ;

  v16bf qa0 = load_frag(qbase + 0,  DKV, lane);
  v16bf qa1 = load_frag(qbase + 32, DKV, lane);

  int row_l = qb + wave * 16 + 8 * hf;   // + j = this lane's global rows
  float mq[8];
#pragma unroll
  for (int j = 0; j < 8; ++j) mq[j] = mrow[row_l + j];

  float m_run[8], l_run[8];
  v8f o0 = {}, o1 = {}, o2 = {}, o3 = {};
#pragma unroll
  for (int j = 0; j < 8; ++j) { m_run[j] = -1e30f; l_run[j] = 0.f; }

  int nchunks = qb / 32 + 4;             // causal: keys < qb+128
  for (int kc = 0; kc < nchunks; ++kc) {
    int key0 = kc * 32;
    __syncthreads();                     // previous chunk's LDS reads done
    {
      // async stage: K chunk (32 keys x 64 dk) + V chunk (64 dv x 32 keys)
      int key = tid >> 3, off = (tid & 7) * 8;   // 2048 halves, 8/thread
      cp16_g2lds(Kc + key * 64 + off, kbase + (size_t)(key0 + key) * DKV + off);
      int dv = tid >> 2, off2 = (tid & 3) * 8;
      cp16_g2lds(Vc + dv * 32 + off2, vbase + (size_t)dv * LSEQ + key0 + off2);
      cp_g2lds_wait();
    }
    __syncthreads();

    bool active = key0 <= (qb + wave * 16 + 15);
    if (active) {
      // --- scores: 2 key tiles x (dk=64 -> 2 WMMAs) ---
      v8f s0 = {}, s1 = {};
      {
        v16bf b;
        b = load_frag(Kc + 0 * 16 * 64 + 0,  64, lane); s0 = WMMA_BF16(qa0, b, s0);
        b = load_frag(Kc + 0 * 16 * 64 + 32, 64, lane); s0 = WMMA_BF16(qa1, b, s0);
        b = load_frag(Kc + 1 * 16 * 64 + 0,  64, lane); s1 = WMMA_BF16(qa0, b, s1);
        b = load_frag(Kc + 1 * 16 * 64 + 32, 64, lane); s1 = WMMA_BF16(qa1, b, s1);
      }
      // --- mask + online softmax ---
      float mk0 = mrow[key0 + c];
      float mk1 = mrow[key0 + 16 + c];
      float scal[8], p0[8], p1[8];
#pragma unroll
      for (int j = 0; j < 8; ++j) {
        int   rg  = row_l + j;
        bool  ms0 = ((key0 + c)      > rg) || (mk0 == 0.f) || (mq[j] == 0.f);
        bool  ms1 = ((key0 + 16 + c) > rg) || (mk1 == 0.f) || (mq[j] == 0.f);
        float v0  = ms0 ? -1e30f : s0[j] * 0.125f;    // 1/sqrt(64)
        float v1  = ms1 ? -1e30f : s1[j] * 0.125f;
        float mx  = fmaxf(v0, v1);
        for (int d = 1; d < 16; d <<= 1) mx = fmaxf(mx, __shfl_xor(mx, d, 32));
        float mn  = fmaxf(m_run[j], mx);
        float pe0 = ms0 ? 0.f : __expf(v0 - mn);
        float pe1 = ms1 ? 0.f : __expf(v1 - mn);
        float ps  = pe0 + pe1;
        for (int d = 1; d < 16; d <<= 1) ps += __shfl_xor(ps, d, 32);
        float sc  = __expf(m_run[j] - mn);
        l_run[j]  = l_run[j] * sc + ps;
        m_run[j]  = mn;
        scal[j]   = sc;
        p0[j] = pe0; p1[j] = pe1;
      }
      // --- transpose P through per-wave LDS (D layout -> A layout) ---
      unsigned short* pw = Pst[wave];
#pragma unroll
      for (int j = 0; j < 8; ++j) {
        int row = 8 * hf + j;
        pw[row * 32 + c]      = f2bf(p0[j]);
        pw[row * 32 + 16 + c] = f2bf(p1[j]);
      }
#pragma unroll
      for (int j = 0; j < 8; ++j) {       // rescale running output
        o0[j] *= scal[j]; o1[j] *= scal[j]; o2[j] *= scal[j]; o3[j] *= scal[j];
      }
      v16bf pa = load_frag(pw, 32, lane); // per-wave DS ops are in-order
      v16bf b0 = load_frag(Vc + 0 * 16 * 32, 32, lane);
      v16bf b1 = load_frag(Vc + 1 * 16 * 32, 32, lane);
      v16bf b2 = load_frag(Vc + 2 * 16 * 32, 32, lane);
      v16bf b3 = load_frag(Vc + 3 * 16 * 32, 32, lane);
      o0 = WMMA_BF16(pa, b0, o0);
      o1 = WMMA_BF16(pa, b1, o1);
      o2 = WMMA_BF16(pa, b2, o2);
      o3 = WMMA_BF16(pa, b3, o3);
    }
  }

  // --- normalize + store context [n][l][h*64 + col] as bf16 ---
  float rl[8];
#pragma unroll
  for (int j = 0; j < 8; ++j) rl[j] = (l_run[j] > 0.f) ? (1.0f / l_run[j]) : 0.f;
  size_t tok = (size_t)n * LSEQ + qb + wave * 16 + 8 * hf;
#pragma unroll
  for (int j = 0; j < 8; ++j) {
    size_t base = (tok + j) * DM + h * DKV + c;
    ctx[base + 0]  = f2bf(o0[j] * rl[j]);
    ctx[base + 16] = f2bf(o1[j] * rl[j]);
    ctx[base + 32] = f2bf(o2[j] * rl[j]);
    ctx[base + 48] = f2bf(o3[j] * rl[j]);
  }
}

// ------------------- output projection -------------------
// grid.x = NL/64, grid.y = DM/256; block = 256 (8 waves).
// Each wave: 32 rows x 64 cols => 8 WMMAs per 6 frag loads.
__global__ void __launch_bounds__(256) k_out(
    const unsigned short* __restrict__ ctx,
    const unsigned short* __restrict__ wot,
    float* __restrict__ out) {
  int mt   = blockIdx.x;
  int ct   = blockIdx.y;
  int wave = threadIdx.x >> 5;
  int lane = threadIdx.x & 31;
  int rt   = wave >> 2;      // 0..1 row sub-tile
  int wc   = wave & 3;       // 0..3 col sub-tile
  const unsigned short* Abase = ctx + (size_t)(mt * 64 + rt * 32) * DM;
  const unsigned short* Bbase = wot + (size_t)(ct * 256 + wc * 64) * DM;

  v8f acc[2][4] = {};
  for (int kk = 0; kk < DM; kk += 32) {
    __builtin_prefetch(Abase + kk + 256, 0, 1);
    __builtin_prefetch(Bbase + kk + 256, 0, 1);
    v16bf a0 = load_frag(Abase + kk, DM, lane);
    v16bf a1 = load_frag(Abase + (size_t)16 * DM + kk, DM, lane);
#pragma unroll
    for (int t = 0; t < 4; ++t) {
      v16bf b = load_frag(Bbase + (size_t)(t * 16) * DM + kk, DM, lane);
      acc[0][t] = WMMA_BF16(a0, b, acc[0][t]);
      acc[1][t] = WMMA_BF16(a1, b, acc[1][t]);
    }
  }
  int c = lane & 15, hf = lane >> 4;
#pragma unroll
  for (int rr = 0; rr < 2; ++rr) {
    size_t row0 = (size_t)mt * 64 + rt * 32 + rr * 16 + 8 * hf;
#pragma unroll
    for (int t = 0; t < 4; ++t) {
      int col = ct * 256 + wc * 64 + t * 16 + c;
#pragma unroll
      for (int j = 0; j < 8; ++j)
        out[(row0 + j) * DM + col] = acc[rr][t][j];
    }
  }
}

// ------------------- host launch -------------------
extern "C" void kernel_launch(void* const* d_in, const int* in_sizes, int n_in,
                              void* d_out, int out_size, void* d_ws, size_t ws_size,
                              hipStream_t stream) {
  const float* x  = (const float*)d_in[0];
  const float* mi = (const float*)d_in[1];
  const float* WQ = (const float*)d_in[2];
  const float* WK = (const float*)d_in[3];
  const float* WV = (const float*)d_in[4];
  const float* WO = (const float*)d_in[5];
  float* out = (float*)d_out;

  // carve workspace (bf16 buffers), 256B aligned
  char* p = (char*)d_ws;
  auto carve = [&](size_t bytes) {
    char* r = p;
    p += (bytes + 255) & ~(size_t)255;
    return (unsigned short*)r;
  };
  unsigned short* xb  = carve((size_t)NL * DM * 2);
  unsigned short* wqt = carve((size_t)HH * DKV * DM * 2);
  unsigned short* wkt = carve((size_t)HH * DKV * DM * 2);
  unsigned short* wvt = carve((size_t)HH * DKV * DM * 2);
  unsigned short* wot = carve((size_t)DM * DM * 2);
  unsigned short* qb  = carve((size_t)HH * NBAT * LSEQ * DKV * 2);
  unsigned short* kb  = carve((size_t)HH * NBAT * LSEQ * DKV * 2);
  unsigned short* vtb = carve((size_t)HH * NBAT * DKV * LSEQ * 2);
  unsigned short* ctx = carve((size_t)NL * DM * 2);

  // 1) precision conversion + weight transposes
  {
    int n = NL * DM;
    k_conv_x<<<(n + 255) / 256, 256, 0, stream>>>(x, xb, n);
  }
  {
    int n = HH * DM * DKV;
    k_conv_w<<<(n + 255) / 256, 256, 0, stream>>>(WQ, wqt);
    k_conv_w<<<(n + 255) / 256, 256, 0, stream>>>(WK, wkt);
    k_conv_w<<<(n + 255) / 256, 256, 0, stream>>>(WV, wvt);
  }
  {
    int n = DM * DM;
    k_conv_wo<<<(n + 255) / 256, 256, 0, stream>>>(WO, wot);
  }

  // 2) QKV projections (V stored transposed)
  k_qkv<<<dim3(NL / 128, 3 * HH), 128, 0, stream>>>(xb, wqt, wkt, wvt, qb, kb, vtb);

  // 3) flash attention -> context (concat heads)
  k_attn<<<dim3(NL / 128, HH), 256, 0, stream>>>(qb, kb, vtb, mi, ctx);

  // 4) output projection -> fp32
  k_out<<<dim3(NL / 64, DM / 256), 256, 0, stream>>>(ctx, wot, out);
}
